// acLSTM_58531814310497
// MI455X (gfx1250) — matrix operand order repaired and checked by reading, not compiled, not run on hardware
//
#include <hip/hip_runtime.h>
#include <hip/hip_bf16.h>

// Problem dims (compile-time, from the reference)
#define B_   64
#define T_   512
#define F_   132
#define H_   1024
#define FP_  160   // input feature dim padded to multiple of 32
#define NDP_ 144   // decoder rows padded to multiple of 16

typedef __attribute__((ext_vector_type(16))) __bf16 v16bf;
typedef __attribute__((ext_vector_type(8)))  __bf16 v8bf;
typedef __attribute__((ext_vector_type(8)))  float  v8f;

// ---------------- device helpers ----------------

__device__ __forceinline__ __bf16 f2bf(float x) {
    union { float f; unsigned u; } v; v.f = x;
    unsigned r = v.u + 0x7FFFu + ((v.u >> 16) & 1u);   // RTNE
    unsigned short hs = (unsigned short)(r >> 16);
    __bf16 b;
    __builtin_memcpy(&b, &hs, 2);
    return b;
}

// A-matrix fragment (16x32 bf16, MxK): lane&15 = M row;
// lanes 0-15 hold K = kk+[0..15]; lanes 16-31 hold kk+[16..31]
__device__ __forceinline__ v16bf ldA(const __bf16* __restrict__ A, int row, int ldk,
                                     int kk, int lane) {
    const int kb = (lane < 16) ? 0 : 16;
    union { v16bf v; v8bf h[2]; } u;
    u.h[0] = *(const v8bf*)(A + (size_t)row * ldk + kk + kb);
    u.h[1] = *(const v8bf*)(A + (size_t)row * ldk + kk + kb + 8);
    return u.v;
}

// B-matrix fragment (32x16 bf16, KxN) for gates = x @ W^T with row-major W(N,K):
// lane&15 = N column (weight row); lanes 0-15 hold K=kk+[0..15], lanes 16-31 kk+[16..31]
// -> 16 contiguous bf16 = one 32-byte load from W's row.
__device__ __forceinline__ v16bf ldB(const __bf16* __restrict__ W, int nrow, int ldk,
                                     int kk, int lane) {
    const int kb = (lane < 16) ? 0 : 16;
    return *(const v16bf*)(W + (size_t)nrow * ldk + kk + kb);
}

__device__ __forceinline__ v8f wmma_bf(v16bf a, v16bf b, v8f c) {
    return __builtin_amdgcn_wmma_f32_16x16x32_bf16(
        /*neg_a=*/false, a, /*neg_b=*/false, b,
        /*c_mod=*/(short)0, c, /*reuse_a=*/false, /*reuse_b=*/false);
}

// Branch-free activations: single v_exp_f32 each, saturate correctly via
// __expf's inf/0 behavior (no ocml range-split exec-mask branches).
__device__ __forceinline__ float sigm(float x) { return 1.0f / (1.0f + __expf(-x)); }
__device__ __forceinline__ float tanh_fast(float x) {
    return 1.0f - 2.0f / (1.0f + __expf(2.0f * x));
}

// ---------------- per-step kernels ----------------

// Fused gates for all 3 LSTM cells (they all consume PREVIOUS hidden states).
// Grid: 3 layers * 64 column-slices = 192 blocks of 256 threads (8 waves).
// Wave = (khalf, gate): khalf 0 -> input-side GEMM, khalf 1 -> recurrent GEMM.
// Each wave owns ONE gate's weight slice (weights read exactly once per WG)
// and computes all 4 batch tiles; partial gate tiles are reduced via LDS and
// the LSTM pointwise runs across all 8 waves.
__global__ __launch_bounds__(256)
void acLSTM_gates(const __bf16* __restrict__ Xpad,    // B_ x FP_
                  const __bf16* __restrict__ hprev,   // 3 x B_ x H_
                  __bf16*       __restrict__ hnew,    // 3 x B_ x H_
                  float*        __restrict__ cst,     // 3 x B_ x H_  (in-place)
                  const __bf16* __restrict__ Wih1p,   // 4H x FP_
                  const __bf16* __restrict__ Whh1,    // 4H x H_
                  const __bf16* __restrict__ Wih2,
                  const __bf16* __restrict__ Whh2,
                  const __bf16* __restrict__ Wih3,
                  const __bf16* __restrict__ Whh3,
                  const float*  __restrict__ bias)    // 3 x 4H  (bih + bhh)
{
    __shared__ float xch[2][4][4][32][8];   // [khalf][gate][mtile][lane][r] = 32 KB

    const int layer = blockIdx.x >> 6;
    const int nbase = (blockIdx.x & 63) << 4;
    const int lane  = threadIdx.x & 31;
    const int wv    = threadIdx.x >> 5;     // 0..7
    const int gate  = wv & 3;
    const int khalf = wv >> 2;
    const int col   = lane & 15;

    const __bf16 *A1, *W1, *A2, *W2;
    int K1;
    if (layer == 0)      { A1 = Xpad;             W1 = Wih1p; K1 = FP_; A2 = hprev;             W2 = Whh1; }
    else if (layer == 1) { A1 = hprev;            W1 = Wih2;  K1 = H_;  A2 = hprev + B_ * H_;   W2 = Whh2; }
    else                 { A1 = hprev + B_ * H_;  W1 = Wih3;  K1 = H_;  A2 = hprev + 2*B_*H_;   W2 = Whh3; }

    // This wave's GEMM: khalf 0 -> (A1,W1,K1); khalf 1 -> (A2,W2,H_).
    const __bf16* A = (khalf == 0) ? A1 : A2;
    const __bf16* W = (khalf == 0) ? W1 : W2;
    int           K = (khalf == 0) ? K1 : H_;
    // K is wave-uniform (khalf is per-wave): force into an SGPR so the K-loop
    // compiles to a scalar s_cmp/s_cbranch instead of exec-masked VALU flow.
    K = __builtin_amdgcn_readfirstlane(K);
    const int ngate = gate * H_ + nbase + col;   // weight row for this gate/lane

    v8f acc[4] = {};   // one accumulator per batch tile (m-tile 0..3)

#pragma unroll 2
    for (int kk = 0; kk < K; kk += 32) {
        v16bf b = ldB(W, ngate, K, kk, lane);                 // loaded ONCE per WG
        acc[0] = wmma_bf(ldA(A,  0 + col, K, kk, lane), b, acc[0]);
        acc[1] = wmma_bf(ldA(A, 16 + col, K, kk, lane), b, acc[1]);
        acc[2] = wmma_bf(ldA(A, 32 + col, K, kk, lane), b, acc[2]);
        acc[3] = wmma_bf(ldA(A, 48 + col, K, kk, lane), b, acc[3]);
    }

    // Exchange partial gate tiles through LDS.
#pragma unroll
    for (int mt = 0; mt < 4; ++mt) {
#pragma unroll
        for (int r = 0; r < 8; ++r)
            xch[khalf][gate][mt][lane][r] = acc[mt][r];
    }
    __syncthreads();

    // Pointwise LSTM across all 8 waves: wave handles (mtile = wv&3, rhalf = wv>>2).
    const int   mt = wv & 3;
    const int   rh = wv >> 2;
    const float* bl = bias + (size_t)layer * 4 * H_;
    const float bi = bl[0*H_ + nbase + col];
    const float bf = bl[1*H_ + nbase + col];
    const float bg = bl[2*H_ + nbase + col];
    const float bo = bl[3*H_ + nbase + col];
    float*  cL = cst  + (size_t)layer * B_ * H_;
    __bf16* hL = hnew + (size_t)layer * B_ * H_;
    const int hid   = nbase + col;
    const int brow0 = mt * 16 + ((lane >= 16) ? 8 : 0);
#pragma unroll
    for (int r4 = 0; r4 < 4; ++r4) {
        const int r   = rh * 4 + r4;
        const int idx = (brow0 + r) * H_ + hid;
        float gi = xch[0][0][mt][lane][r] + xch[1][0][mt][lane][r];
        float gf = xch[0][1][mt][lane][r] + xch[1][1][mt][lane][r];
        float gg = xch[0][2][mt][lane][r] + xch[1][2][mt][lane][r];
        float go = xch[0][3][mt][lane][r] + xch[1][3][mt][lane][r];
        float iv = sigm(gi + bi);
        float fv = sigm(gf + bf);
        float gv = tanh_fast(gg + bg);
        float ov = sigm(go + bo);
        float cn = fv * cL[idx] + iv * gv;
        cL[idx] = cn;
        hL[idx] = f2bf(ov * tanh_fast(cn));
    }
}

// Decoder: out = h2_new @ Wd^T + bd, plus fused teacher-forcing build of the
// NEXT step's padded bf16 input tile. Grid: 9 blocks (NDP_/16) of 128 threads.
__global__ __launch_bounds__(128)
void acLSTM_decode(const __bf16* __restrict__ h2,     // B_ x H_ (new h2)
                   const __bf16* __restrict__ Wdp,    // NDP_ x H_ (rows >=F_ zero)
                   const float*  __restrict__ bd,     // F_
                   float*        __restrict__ out,    // B_ x (T_*F_)
                   const float*  __restrict__ seq,    // B_ x T_ x F_
                   __bf16*       __restrict__ Xpad,   // B_ x FP_
                   int t, int flag_next)
{
    const int nbase = blockIdx.x << 4;
    const int lane  = threadIdx.x & 31;
    const int wave  = threadIdx.x >> 5;
    const int mbase = wave << 4;
    const int col   = lane & 15;
    const int arow  = mbase + col;

    v8f acc = {};
#pragma unroll 2
    for (int kk = 0; kk < H_; kk += 32) {
        v16bf a = ldA(h2, arow, H_, kk, lane);
        v16bf b = ldB(Wdp, nbase + col, H_, kk, lane);
        acc = wmma_bf(a, b, acc);
    }

    const int n = nbase + col;
    if (n < F_) {
        const float bb = bd[n];
        const int brow0 = mbase + ((lane >= 16) ? 8 : 0);
#pragma unroll
        for (int r = 0; r < 8; ++r) {
            const int batch = brow0 + r;
            const float v = acc[r] + bb;
            out[(size_t)batch * (T_ * F_) + (size_t)t * F_ + n] = v;
            if (flag_next >= 0) {
                float xv = flag_next ? seq[((size_t)batch * T_ + (t + 1)) * F_ + n] : v;
                Xpad[batch * FP_ + n] = f2bf(xv);
            }
        }
    }
}

// ---------------- one-time setup kernels ----------------

__global__ void cvt_bf16_k(const float* __restrict__ s, __bf16* __restrict__ d, int n) {
    int i = blockIdx.x * blockDim.x + threadIdx.x;
    if (i < n) d[i] = f2bf(s[i]);
}

__global__ void cvt_pad_k(const float* __restrict__ s, __bf16* __restrict__ d,
                          int srows, int scols, int drows, int dcols) {
    int i = blockIdx.x * blockDim.x + threadIdx.x;
    int n = drows * dcols;
    if (i < n) {
        int r = i / dcols, c = i % dcols;
        d[i] = (r < srows && c < scols) ? f2bf(s[(size_t)r * scols + c]) : f2bf(0.0f);
    }
}

__global__ void bias_k(const float* __restrict__ a, const float* __restrict__ b,
                       float* __restrict__ d, int n) {
    int i = blockIdx.x * blockDim.x + threadIdx.x;
    if (i < n) d[i] = a[i] + b[i];
}

__global__ void init_state_k(float* __restrict__ c, __bf16* __restrict__ h0,
                             __bf16* __restrict__ h1, int n) {
    int i = blockIdx.x * blockDim.x + threadIdx.x;
    if (i < n) { c[i] = 0.0f; h0[i] = f2bf(0.0f); h1[i] = f2bf(0.0f); }
}

__global__ void init_xpad_k(const float* __restrict__ seq, __bf16* __restrict__ x) {
    int i = blockIdx.x * blockDim.x + threadIdx.x;
    if (i < B_ * FP_) {
        int b = i / FP_, c = i % FP_;
        // flag(t=0) == 1 -> teacher-forced frame 0
        x[i] = (c < F_) ? f2bf(seq[(size_t)b * T_ * F_ + c]) : f2bf(0.0f);
    }
}

// ---------------- launcher ----------------

extern "C" void kernel_launch(void* const* d_in, const int* in_sizes, int n_in,
                              void* d_out, int out_size, void* d_ws, size_t ws_size,
                              hipStream_t stream) {
    const float* real_seq = (const float*)d_in[0];
    const float* Wih1 = (const float*)d_in[1];
    const float* Whh1 = (const float*)d_in[2];
    const float* bih1 = (const float*)d_in[3];
    const float* bhh1 = (const float*)d_in[4];
    const float* Wih2 = (const float*)d_in[5];
    const float* Whh2 = (const float*)d_in[6];
    const float* bih2 = (const float*)d_in[7];
    const float* bhh2 = (const float*)d_in[8];
    const float* Wih3 = (const float*)d_in[9];
    const float* Whh3 = (const float*)d_in[10];
    const float* bih3 = (const float*)d_in[11];
    const float* bhh3 = (const float*)d_in[12];
    const float* Wd   = (const float*)d_in[13];
    const float* bd   = (const float*)d_in[14];
    float* out = (float*)d_out;

    // ---- carve workspace (all 256B aligned) ----
    char* w = (char*)d_ws;
    size_t off = 0;
    auto carve = [&](size_t bytes) -> void* {
        void* p = w + off;
        off += (bytes + 255) & ~(size_t)255;
        return p;
    };
    __bf16* wih1p = (__bf16*)carve((size_t)4 * H_ * FP_ * 2);
    __bf16* whh1b = (__bf16*)carve((size_t)4 * H_ * H_ * 2);
    __bf16* wih2b = (__bf16*)carve((size_t)4 * H_ * H_ * 2);
    __bf16* whh2b = (__bf16*)carve((size_t)4 * H_ * H_ * 2);
    __bf16* wih3b = (__bf16*)carve((size_t)4 * H_ * H_ * 2);
    __bf16* whh3b = (__bf16*)carve((size_t)4 * H_ * H_ * 2);
    __bf16* wdp   = (__bf16*)carve((size_t)NDP_ * H_ * 2);
    float*  biasb = (float*)carve((size_t)3 * 4 * H_ * 4);
    __bf16* hbuf0 = (__bf16*)carve((size_t)3 * B_ * H_ * 2);
    __bf16* hbuf1 = (__bf16*)carve((size_t)3 * B_ * H_ * 2);
    float*  cbuf  = (float*)carve((size_t)3 * B_ * H_ * 4);
    __bf16* xpad  = (__bf16*)carve((size_t)B_ * FP_ * 2);
    (void)ws_size; (void)in_sizes; (void)n_in; (void)out_size;

    // ---- one-time conversions / init ----
    const int TB = 256;
    {
        int n = 4 * H_ * FP_;
        cvt_pad_k<<<(n + TB - 1) / TB, TB, 0, stream>>>(Wih1, wih1p, 4 * H_, F_, 4 * H_, FP_);
    }
    {
        int n = 4 * H_ * H_;
        int g = (n + TB - 1) / TB;
        cvt_bf16_k<<<g, TB, 0, stream>>>(Whh1, whh1b, n);
        cvt_bf16_k<<<g, TB, 0, stream>>>(Wih2, wih2b, n);
        cvt_bf16_k<<<g, TB, 0, stream>>>(Whh2, whh2b, n);
        cvt_bf16_k<<<g, TB, 0, stream>>>(Wih3, wih3b, n);
        cvt_bf16_k<<<g, TB, 0, stream>>>(Whh3, whh3b, n);
    }
    {
        int n = NDP_ * H_;
        cvt_pad_k<<<(n + TB - 1) / TB, TB, 0, stream>>>(Wd, wdp, F_, H_, NDP_, H_);
    }
    {
        int n = 4 * H_;
        int g = (n + TB - 1) / TB;
        bias_k<<<g, TB, 0, stream>>>(bih1, bhh1, biasb + 0 * 4 * H_, n);
        bias_k<<<g, TB, 0, stream>>>(bih2, bhh2, biasb + 1 * 4 * H_, n);
        bias_k<<<g, TB, 0, stream>>>(bih3, bhh3, biasb + 2 * 4 * H_, n);
    }
    {
        int n = 3 * B_ * H_;
        init_state_k<<<(n + TB - 1) / TB, TB, 0, stream>>>(cbuf, hbuf0, hbuf1, n);
        int m = B_ * FP_;
        init_xpad_k<<<(m + TB - 1) / TB, TB, 0, stream>>>(real_seq, xpad);
    }

    // ---- sequential timestep loop (graph-captured; 2 kernels/step) ----
    __bf16* hb[2] = {hbuf0, hbuf1};
    for (int t = 0; t < T_; ++t) {
        __bf16* hprev = hb[t & 1];
        __bf16* hnew  = hb[(t + 1) & 1];

        acLSTM_gates<<<192, 256, 0, stream>>>(
            xpad, hprev, hnew, cbuf,
            wih1p, whh1b, wih2b, whh2b, wih3b, whh3b, biasb);

        // teacher-forcing schedule: 5 ground-truth frames then 5 free-running
        int flag_next = (t + 1 < T_) ? ((((t + 1) % 10) < 5) ? 1 : 0) : -1;
        acLSTM_decode<<<NDP_ / 16, 128, 0, stream>>>(
            hnew + 2 * B_ * H_, wdp, bd, out, real_seq, xpad, t, flag_next);
    }
}
